// MultiheadPointTransformerLayer_83219286327742
// MI455X (gfx1250) — compile-verified, matrix operand order checked by reading, not run on hardware
//
#include <hip/hip_runtime.h>
#include <hip/hip_bf16.h>
#include <float.h>

// ---------------- problem constants ----------------
#define BATCH  2
#define NPTS   4096
#define EMB    256
#define HEADS  4
#define DH     64
#define INNER  256     // HEADS*DH
#define KNN    16
#define PH     64      // pos mlp hidden
#define AH     256     // attn mlp hidden (per head)
#define LDP    72      // padded f32 row stride (keeps 32B alignment, offsets banks)

// ---------------- vector types ----------------
typedef __attribute__((ext_vector_type(16))) __bf16 v16bf;
typedef __attribute__((ext_vector_type(8)))  __bf16 v8bf;
typedef __attribute__((ext_vector_type(4)))  __bf16 v4bf;
typedef __attribute__((ext_vector_type(8)))  float  v8f;
typedef __attribute__((ext_vector_type(4)))  unsigned int v4u;
typedef __attribute__((ext_vector_type(4)))  int v4i;
typedef __attribute__((ext_vector_type(8)))  int v8i;

#if __has_builtin(__builtin_amdgcn_tensor_load_to_lds)
#define HAVE_TDM 1
#else
#define HAVE_TDM 0
#endif

__device__ __forceinline__ v8f wmma_bf16(v16bf a, v16bf b, v8f c) {
  return __builtin_amdgcn_wmma_f32_16x16x32_bf16(
      false, a, false, b, (short)0, c, false, false);
}

// One loader serves both operands (row-major [M][K] for A, transposed [N][K]
// for B). Per-lane data is two contiguous 8-element runs -> 2x ds_load_b128.
__device__ __forceinline__ v16bf load_frag_16x32(const __bf16* lds, int maj0, int k0, int ld) {
  int lane = threadIdx.x & 31;
  int half = lane >> 4, m = lane & 15;
  const __bf16* p = lds + (size_t)(maj0 + m) * ld + k0;
  v8bf lo = *(const v8bf*)(p + half * 8);
  v8bf hi = *(const v8bf*)(p + 16 + half * 8);
  return __builtin_shufflevector(lo, hi, 0, 1, 2, 3, 4, 5, 6, 7,
                                 8, 9, 10, 11, 12, 13, 14, 15);
}

#if HAVE_TDM
// Issue one TDM 2D tile load (global -> LDS). All sizes in 8-byte units
// (D# data_size = 3). Descriptor bitfields per CDNA5 ISA ch.8.
__device__ __forceinline__ void tdm_load_2d(unsigned int lds_off, const void* gptr,
                                            unsigned int tile_d0, unsigned int tile_d1,
                                            unsigned int row_stride_u) {
  unsigned long long ga = (unsigned long long)gptr;
  v4u g0;
  g0.x = 1u;                                               // count=1 (valid), user mode
  g0.y = lds_off;                                          // lds_addr (bytes)
  g0.z = (unsigned int)ga;                                 // global_addr[31:0]
  g0.w = (unsigned int)((ga >> 32) & 0x01FFFFFFu) | (2u << 30); // addr[56:32], type=2
  v8i g1;
  g1[0] = (int)(3u << 16);                                 // wg_mask=0, data_size=8B
  g1[1] = (int)((row_stride_u & 0xFFFFu) << 16);           // tensor_dim0[15:0]
  g1[2] = (int)((row_stride_u >> 16) | ((tile_d1 & 0xFFFFu) << 16)); // dim0 hi | dim1 lo
  g1[3] = (int)((tile_d1 >> 16) | (tile_d0 << 16));        // dim1 hi | tile_dim0
  g1[4] = (int)(tile_d1 & 0xFFFFu);                        // tile_dim1 (tile_dim2=0)
  g1[5] = (int)row_stride_u;                               // tensor_dim0_stride[31:0]
  g1[6] = 0; g1[7] = 0;
  v4i gz = {0, 0, 0, 0};
#if defined(__clang_major__) && (__clang_major__ >= 23)
  v8i gz8 = {0, 0, 0, 0, 0, 0, 0, 0};
  __builtin_amdgcn_tensor_load_to_lds(g0, g1, gz, gz, gz8, 0);
#else
  __builtin_amdgcn_tensor_load_to_lds(g0, g1, gz, gz, 0);
#endif
}
#endif

// ---------------- kernel 0: weight prep (fp32 -> bf16, transposed) ----------
__global__ void __launch_bounds__(256)
prep_weights_kernel(const float* __restrict__ Wp2, const float* __restrict__ Wa1,
                    const float* __restrict__ Wa2, __bf16* __restrict__ wp2T,
                    __bf16* __restrict__ wa1T, __bf16* __restrict__ wa2T) {
  int i = blockIdx.x * 256 + threadIdx.x;
  if (i < HEADS * DH * PH) {                  // wp2T[h][d][k] = Wp2[k][h*64+d]
    int h = i >> 12, d = (i >> 6) & 63, k = i & 63;
    wp2T[i] = (__bf16)Wp2[(size_t)k * INNER + h * DH + d];
  }
  if (i < HEADS * AH * DH) {                  // wa1T[h][e][d] = Wa1[h][d][e]
    int h = i / (AH * DH), e = (i / DH) % AH, d = i % DH;
    wa1T[i] = (__bf16)Wa1[((size_t)h * DH + d) * AH + e];
  }
  if (i < HEADS * DH * AH) {                  // wa2T[h][d][e] = Wa2[h][e][d]
    int h = i / (DH * AH), d = (i / AH) % DH, e = i % AH;
    wa2T[i] = (__bf16)Wa2[((size_t)h * AH + e) * DH + d];
  }
}

// ---------------- kernel 1: brute-force kNN (top-16 by squared distance) ----
__global__ void __launch_bounds__(128)
knn_kernel(const float* __restrict__ pos, int* __restrict__ idx_out) {
  __shared__ float sp[256 * 3];
  int bb = blockIdx.x / (NPTS / 128);
  int i  = (blockIdx.x % (NPTS / 128)) * 128 + threadIdx.x;
  const float* pb = pos + (size_t)bb * NPTS * 3;
  float px = pb[i * 3 + 0], py = pb[i * 3 + 1], pz = pb[i * 3 + 2];

  float bd[KNN]; int bi[KNN];
#pragma unroll
  for (int k = 0; k < KNN; ++k) { bd[k] = FLT_MAX; bi[k] = 0; }
  float worst = FLT_MAX;

  for (int j0 = 0; j0 < NPTS; j0 += 256) {
    for (int u = threadIdx.x; u < 256 * 3; u += 128) sp[u] = pb[(size_t)j0 * 3 + u];
    __syncthreads();
    for (int jj = 0; jj < 256; ++jj) {
      float dx = px - sp[jj * 3 + 0];
      float dy = py - sp[jj * 3 + 1];
      float dz = pz - sp[jj * 3 + 2];
      float d2 = dx * dx + dy * dy + dz * dz;
      if (d2 < worst) {
        int ws = 0; float wv = bd[0];
#pragma unroll
        for (int k = 1; k < KNN; ++k) if (bd[k] > wv) { wv = bd[k]; ws = k; }
        bd[ws] = d2; bi[ws] = j0 + jj;
        wv = bd[0];
#pragma unroll
        for (int k = 1; k < KNN; ++k) wv = fmaxf(wv, bd[k]);
        worst = wv;
      }
    }
    __syncthreads();
  }
  int* out = idx_out + ((size_t)bb * NPTS + i) * KNN;
#pragma unroll
  for (int k = 0; k < KNN; ++k) out[k] = bi[k];
}

// ---------------- kernel 2/5: generic WMMA GEMM, C = A @ W (+epilogue) ------
template <int MODE>   // 1: qkv scatter, 2: add bias[col]
__global__ void __launch_bounds__(256)
gemm_wmma_kernel(const float* __restrict__ A, const float* __restrict__ W,
                 const float* __restrict__ bias, float* __restrict__ C,
                 int M, int N, int Kd) {
  __shared__ __bf16 sA[128][32];    // row-major [m][k]
  __shared__ __bf16 sBt[32][32];    // transposed [n][k]
  int bm = blockIdx.y * 128, bn = blockIdx.x * 32;
  int t = threadIdx.x, wave = t >> 5, lane = t & 31;
  v8f acc0 = {}, acc1 = {};

  for (int k0 = 0; k0 < Kd; k0 += 32) {
    for (int u = t; u < 128 * 8; u += 256) {
      int r = u >> 3, c4 = (u & 7) * 4;
      float4 f = *(const float4*)&A[(size_t)(bm + r) * Kd + k0 + c4];
      v4bf hh; hh[0] = (__bf16)f.x; hh[1] = (__bf16)f.y; hh[2] = (__bf16)f.z; hh[3] = (__bf16)f.w;
      *(v4bf*)&sA[r][c4] = hh;
    }
    for (int u = t; u < 32 * 8; u += 256) {
      int r = u >> 3, c4 = (u & 7) * 4;   // r = k, c4 = n
      float4 f = *(const float4*)&W[(size_t)(k0 + r) * N + bn + c4];
      sBt[c4 + 0][r] = (__bf16)f.x; sBt[c4 + 1][r] = (__bf16)f.y;
      sBt[c4 + 2][r] = (__bf16)f.z; sBt[c4 + 3][r] = (__bf16)f.w;
    }
    if (k0 + 32 < Kd)
      __builtin_prefetch(&A[(size_t)(bm + (t >> 1)) * Kd + k0 + 32], 0, 1);
    __syncthreads();
    v16bf a = load_frag_16x32(&sA[0][0], wave * 16, 0, 32);
    acc0 = wmma_bf16(a, load_frag_16x32(&sBt[0][0], 0,  0, 32), acc0);
    acc1 = wmma_bf16(a, load_frag_16x32(&sBt[0][0], 16, 0, 32), acc1);
    __syncthreads();
  }

  int halfw = lane >> 4, n = lane & 15;
#pragma unroll
  for (int r8 = 0; r8 < 8; ++r8) {
    int row = bm + wave * 16 + halfw * 8 + r8;
#pragma unroll
    for (int s = 0; s < 2; ++s) {
      int col = bn + s * 16 + n;
      float v = (s == 0) ? acc0[r8] : acc1[r8];
      if (MODE == 2) v += bias[col];
      if (MODE == 1) {
        int part = col >> 8;
        int cc = col & 255;
        int hh = cc >> 6, dd = cc & 63;
        int b2 = row / NPTS, ii = row % NPTS;
        size_t plane = (size_t)NPTS * DH;
        C[(size_t)part * (BATCH * HEADS) * plane +
          (((size_t)b2 * HEADS + hh) * NPTS + ii) * DH + dd] = v;
      } else {
        C[(size_t)row * N + col] = v;
      }
    }
  }
}

// ---------------- kernel 3: fused rpe + attn-MLP + softmax (pass 1) --------
// LDS pool with explicit offsets so the TDM descriptors can address it.
#define OFF_SW    0                                // float [64][72]  18432B
#define OFF_SSIM  18432                            // float [64][72]  18432B
#define OFF_SAIN  36864                            // bf16  [64][64]   8192B
#define OFF_SHID  45056                            // bf16  [64][32]   4096B
#define OFF_WPOOL 49152                            // bf16 Wp2T/Wa1T   8192/4096B
#define OFF_WA2   53248                            // bf16 Wa2T chunk  4096B
#define OFF_SSQ   57344                            // float [16][64]   4096B
#define OFF_SIDX  61440                            // int [64]          256B
#define SMEM_SZ   61696

__global__ void __launch_bounds__(256)
attn_pass1_kernel(const float* __restrict__ pos, const unsigned char* __restrict__ mask,
                  const int* __restrict__ idx, const float* __restrict__ qkv,
                  const float* __restrict__ Wp1, const float* __restrict__ bp1,
                  const __bf16* __restrict__ wp2T_g, const float* __restrict__ bp2,
                  const __bf16* __restrict__ wa1T_g, const float* __restrict__ ba1,
                  const __bf16* __restrict__ wa2T_g, const float* __restrict__ ba2,
                  float* __restrict__ P, float* __restrict__ sumsq) {
  const int R = 64;
  __shared__ __align__(16) unsigned char smem[SMEM_SZ];
  float  (*sW)[LDP]   = (float(*)[LDP])(smem + OFF_SW);
  float  (*sSim)[LDP] = (float(*)[LDP])(smem + OFF_SSIM);
  __bf16 (*sAin)[DH]  = (__bf16(*)[DH])(smem + OFF_SAIN);
  __bf16 (*sHid)[32]  = (__bf16(*)[32])(smem + OFF_SHID);
  __bf16* sWp2T = (__bf16*)(smem + OFF_WPOOL);     // [64 d][64 k]
  __bf16* sWa1T = (__bf16*)(smem + OFF_WPOOL);     // [32 e][64 d] per chunk
  __bf16* sWa2T = (__bf16*)(smem + OFF_WA2);       // [64 d][32 e] per chunk
  float  (*sSq)[DH]   = (float(*)[DH])(smem + OFF_SSQ);
  int*    sIdx        = (int*)(smem + OFF_SIDX);

  const int bh = blockIdx.y;
  const int h  = bh % HEADS;
  const int bb = bh / HEADS;
  const int i0 = blockIdx.x * 4;
  const int t  = threadIdx.x;
  const int wave = t >> 5, lane = t & 31;
  const int halfw = lane >> 4, ln = lane & 15;

  const float* pb = pos + (size_t)bb * NPTS * 3;
  const size_t plane = (size_t)NPTS * DH;
  const float* qbase = qkv + (size_t)bh * plane;
  const float* kbase = qkv + (size_t)BATCH * HEADS * plane + (size_t)bh * plane;
  const float* vbase = qkv + 2 * (size_t)BATCH * HEADS * plane + (size_t)bh * plane;

  if (t < R) sIdx[t] = idx[((size_t)bb * NPTS + (i0 + t / KNN)) * KNN + (t % KNN)];
  for (int u = t; u < KNN * DH; u += 256) (&sSq[0][0])[u] = 0.f;
  // stage Wp2h^T (8KB contiguous) via Tensor Data Mover
#if HAVE_TDM
  if (wave == 0) {
    tdm_load_2d(OFF_WPOOL, wp2T_g + (size_t)h * DH * PH, 1024, 1, 1024);
    __builtin_amdgcn_s_wait_tensorcnt(0);
  }
#else
  for (int u = t; u < DH * PH; u += 256) sWp2T[u] = wp2T_g[(size_t)h * DH * PH + u];
#endif
  __syncthreads();

  // ---- pos-MLP layer 1 (VALU fp32): hid1 = relu(rel @ Wp1 + bp1) ----
  {
    int r = t >> 2, cg = t & 3;
    int qi = i0 + r / KNN, j = sIdx[r];
    float rx = pb[j * 3 + 0] - pb[qi * 3 + 0];
    float ry = pb[j * 3 + 1] - pb[qi * 3 + 1];
    float rz = pb[j * 3 + 2] - pb[qi * 3 + 2];
#pragma unroll
    for (int g = 0; g < 2; ++g) {
      v8bf hv;
#pragma unroll
      for (int e = 0; e < 8; ++e) {
        int c = cg * 16 + g * 8 + e;
        float v = rx * Wp1[c] + ry * Wp1[PH + c] + rz * Wp1[2 * PH + c] + bp1[c];
        hv[e] = (__bf16)fmaxf(v, 0.f);
      }
      *(v8bf*)&sAin[r][cg * 16 + g * 8] = hv;
    }
  }
  __syncthreads();

  // ---- rpe^T = Wp2h^T @ hid1^T ----
  {
    int tr = wave >> 1;
    int tc0 = (wave & 1) * 2;
    v8f acc0 = {}, acc1 = {};
#pragma unroll
    for (int k0 = 0; k0 < PH; k0 += 32) {
      v16bf a = load_frag_16x32(sWp2T, tr * 16, k0, 64);
      acc0 = wmma_bf16(a, load_frag_16x32(&sAin[0][0], tc0 * 16,      k0, DH), acc0);
      acc1 = wmma_bf16(a, load_frag_16x32(&sAin[0][0], tc0 * 16 + 16, k0, DH), acc1);
    }
    v8f o0, o1;
#pragma unroll
    for (int r8 = 0; r8 < 8; ++r8) {
      float b = bp2[h * DH + tr * 16 + halfw * 8 + r8];
      o0[r8] = acc0[r8] + b;
      o1[r8] = acc1[r8] + b;
    }
    *(v8f*)&sW[tc0 * 16 + ln][tr * 16 + halfw * 8]      = o0;
    *(v8f*)&sW[tc0 * 16 + 16 + ln][tr * 16 + halfw * 8] = o1;
  }
  __syncthreads();

  // ---- attn_in = q_i - k_j + rpe (bf16), w = v_j + rpe (f32) ----
  {
    int r = t >> 2, cg = t & 3;
    int qi = i0 + r / KNN, j = sIdx[r];
    const float* qr = qbase + (size_t)qi * DH + cg * 16;
    const float* kr = kbase + (size_t)j * DH + cg * 16;
    const float* vr = vbase + (size_t)j * DH + cg * 16;
    float* wrow = &sW[r][cg * 16];
    __bf16* arow = &sAin[r][cg * 16];
#pragma unroll
    for (int g = 0; g < 2; ++g) {
      v8f rp = *(const v8f*)(wrow + g * 8);
      v8bf ab; v8f wv;
#pragma unroll
      for (int e = 0; e < 8; ++e) {
        float rpv = rp[e];
        ab[e] = (__bf16)(qr[g * 8 + e] - kr[g * 8 + e] + rpv);
        wv[e] = vr[g * 8 + e] + rpv;
      }
      *(v8bf*)(arow + g * 8) = ab;
      *(v8f*)(wrow + g * 8) = wv;
    }
  }
  __syncthreads();

  // ---- attn MLP chunked over hidden: sim^T += Wa2c^T @ relu(Wa1c^T @ Ain^T) ----
  v8f sim0 = {}, sim1 = {};
  const int str  = wave >> 1;
  const int stc0 = (wave & 1) * 2;
  for (int e0 = 0; e0 < AH; e0 += 32) {
#if HAVE_TDM
    if (wave == 0) {
      // Wa1^T chunk: contiguous 4KB; Wa2^T chunk: 2D tile 64 rows x 64B, stride 512B
      tdm_load_2d(OFF_WPOOL, wa1T_g + ((size_t)h * AH + e0) * DH, 512, 1, 512);
      tdm_load_2d(OFF_WA2,   wa2T_g + (size_t)h * DH * AH + e0, 8, 64, (AH * 2) / 8);
      __builtin_amdgcn_s_wait_tensorcnt(0);
    }
#else
    for (int u = t; u < 32 * 64; u += 256)
      sWa1T[u] = wa1T_g[((size_t)h * AH + e0) * DH + u];
    for (int u = t; u < 64 * 32; u += 256) {
      int d = u >> 5, e = u & 31;
      sWa2T[u] = wa2T_g[((size_t)h * DH + d) * AH + e0 + e];
    }
#endif
    __syncthreads();
    {   // hid^T chunk [32 e][64 row]: 8 tiles, 1 per wave
      int tre = wave >> 2, trc = wave & 3;
      v8f acc = {};
      acc = wmma_bf16(load_frag_16x32(sWa1T, tre * 16, 0,  64),
                      load_frag_16x32(&sAin[0][0], trc * 16, 0,  DH), acc);
      acc = wmma_bf16(load_frag_16x32(sWa1T, tre * 16, 32, 64),
                      load_frag_16x32(&sAin[0][0], trc * 16, 32, DH), acc);
      v8bf hv;
#pragma unroll
      for (int r8 = 0; r8 < 8; ++r8) {
        float v = acc[r8] + ba1[(size_t)h * AH + e0 + tre * 16 + halfw * 8 + r8];
        hv[r8] = (__bf16)fmaxf(v, 0.f);
      }
      *(v8bf*)&sHid[trc * 16 + ln][tre * 16 + halfw * 8] = hv;
    }
    __syncthreads();
    {   // sim^T [64 d][64 row] += Wa2c^T [d][32 e] @ hid^T [e][row]
      v16bf a = load_frag_16x32(sWa2T, str * 16, 0, 32);
      sim0 = wmma_bf16(a, load_frag_16x32(&sHid[0][0], stc0 * 16,      0, 32), sim0);
      sim1 = wmma_bf16(a, load_frag_16x32(&sHid[0][0], stc0 * 16 + 16, 0, 32), sim1);
    }
    __syncthreads();
  }

  // ---- sim epilogue: + ba2, pair mask, vector store to sSim[row][d] ----
  {
    const unsigned char* mb = mask + (size_t)bb * NPTS;
    int row0 = stc0 * 16 + ln, row1 = row0 + 16;
    int qi0 = i0 + row0 / KNN, j0 = sIdx[row0];
    int qi1 = i0 + row1 / KNN, j1 = sIdx[row1];
    bool ok0 = mb[qi0] && mb[j0];
    bool ok1 = mb[qi1] && mb[j1];
    v8f o0, o1;
#pragma unroll
    for (int r8 = 0; r8 < 8; ++r8) {
      float b = ba2[(size_t)h * DH + str * 16 + halfw * 8 + r8];
      o0[r8] = ok0 ? (sim0[r8] + b) : -FLT_MAX;
      o1[r8] = ok1 ? (sim1[r8] + b) : -FLT_MAX;
    }
    *(v8f*)&sSim[row0][str * 16 + halfw * 8] = o0;
    *(v8f*)&sSim[row1][str * 16 + halfw * 8] = o1;
  }
  __syncthreads();

  // ---- softmax over K per (query,d); write P = attn*w; accumulate attn^2 ----
  {
    int ql = t >> 6, d = t & 63;
    float m = -FLT_MAX;
#pragma unroll
    for (int k = 0; k < KNN; ++k) m = fmaxf(m, sSim[ql * KNN + k][d]);
    float e[KNN]; float s = 0.f;
#pragma unroll
    for (int k = 0; k < KNN; ++k) { e[k] = __expf(sSim[ql * KNN + k][d] - m); s += e[k]; }
    float inv = 1.f / s;
    size_t pbase = (((size_t)bh * NPTS + (i0 + ql)) * KNN) * DH + d;
#pragma unroll
    for (int k = 0; k < KNN; ++k) {
      float a = e[k] * inv;
      atomicAdd(&sSq[k][d], a * a);
      P[pbase + (size_t)k * DH] = a * sW[ql * KNN + k][d];
    }
  }
  __syncthreads();
  for (int u = t; u < KNN * DH; u += 256)
    atomicAdd(&sumsq[(size_t)bh * KNN * DH + u], (&sSq[0][0])[u]);
}

// ---------------- kernel 4: column-normalize and reduce over neighbors -----
__global__ void __launch_bounds__(256)
agg_kernel(const float* __restrict__ P, const float* __restrict__ sumsq,
           float* __restrict__ agg) {
  __shared__ float sInv[KNN * DH];
  int bh = blockIdx.y;
  int bb = bh / HEADS, h = bh % HEADS;
  int i0 = blockIdx.x * 4;
  for (int u = threadIdx.x; u < KNN * DH; u += 256) {
    float nrm = sqrtf(sumsq[(size_t)bh * KNN * DH + u]);
    sInv[u] = 1.f / fmaxf(nrm, 1e-12f);
  }
  __syncthreads();
  int ql = threadIdx.x >> 6, d = threadIdx.x & 63;
  int i = i0 + ql;
  const float* pr = P + (((size_t)bh * NPTS + i) * KNN) * DH + d;
  float acc = 0.f;
#pragma unroll
  for (int k = 0; k < KNN; ++k) acc += pr[(size_t)k * DH] * sInv[k * DH + d];
  agg[((size_t)bb * NPTS + i) * INNER + h * DH + d] = acc;
}

__global__ void zero_kernel(float* __restrict__ p, int n) {
  int i = blockIdx.x * 256 + threadIdx.x;
  if (i < n) p[i] = 0.f;
}

// ---------------- host launcher ----------------
extern "C" void kernel_launch(void* const* d_in, const int* in_sizes, int n_in,
                              void* d_out, int out_size, void* d_ws, size_t ws_size,
                              hipStream_t stream) {
  (void)in_sizes; (void)n_in; (void)out_size; (void)ws_size;
  const float* x    = (const float*)d_in[0];
  const unsigned char* mask = (const unsigned char*)d_in[1];
  const float* pos  = (const float*)d_in[2];
  const float* Wqkv = (const float*)d_in[3];
  const float* Wp1  = (const float*)d_in[4];
  const float* bp1  = (const float*)d_in[5];
  const float* Wp2  = (const float*)d_in[6];
  const float* bp2  = (const float*)d_in[7];
  const float* Wa1  = (const float*)d_in[8];
  const float* ba1  = (const float*)d_in[9];
  const float* Wa2  = (const float*)d_in[10];
  const float* ba2  = (const float*)d_in[11];
  const float* Wo   = (const float*)d_in[12];
  const float* bo   = (const float*)d_in[13];
  float* out = (float*)d_out;

  char* ws = (char*)d_ws;
  size_t off = 0;
  auto carve = [&](size_t bytes) { void* p = ws + off; off = (off + bytes + 255) & ~(size_t)255; return p; };
  int*    idx   = (int*)   carve((size_t)BATCH * NPTS * KNN * 4);
  float*  qkv   = (float*) carve((size_t)3 * BATCH * HEADS * NPTS * DH * 4);
  float*  P     = (float*) carve((size_t)BATCH * HEADS * NPTS * KNN * DH * 4);
  float*  sumsq = (float*) carve((size_t)BATCH * HEADS * KNN * DH * 4);
  float*  agg   = (float*) carve((size_t)BATCH * NPTS * INNER * 4);
  __bf16* wp2T  = (__bf16*)carve((size_t)HEADS * DH * PH * 2);
  __bf16* wa1T  = (__bf16*)carve((size_t)HEADS * AH * DH * 2);
  __bf16* wa2T  = (__bf16*)carve((size_t)HEADS * DH * AH * 2);

  zero_kernel<<<(BATCH * HEADS * KNN * DH + 255) / 256, 256, 0, stream>>>(
      sumsq, BATCH * HEADS * KNN * DH);

  prep_weights_kernel<<<(HEADS * AH * DH + 255) / 256, 256, 0, stream>>>(
      Wp2, Wa1, Wa2, wp2T, wa1T, wa2T);

  knn_kernel<<<BATCH * (NPTS / 128), 128, 0, stream>>>(pos, idx);

  {
    dim3 g((3 * INNER) / 32, (BATCH * NPTS) / 128);
    gemm_wmma_kernel<1><<<g, 256, 0, stream>>>(x, Wqkv, nullptr, qkv,
                                               BATCH * NPTS, 3 * INNER, EMB);
  }
  {
    dim3 g(NPTS / 4, BATCH * HEADS);
    attn_pass1_kernel<<<g, 256, 0, stream>>>(pos, mask, idx, qkv,
                                             Wp1, bp1, wp2T, bp2,
                                             wa1T, ba1, wa2T, ba2, P, sumsq);
  }
  {
    dim3 g(NPTS / 4, BATCH * HEADS);
    agg_kernel<<<g, 256, 0, stream>>>(P, sumsq, agg);
  }
  {
    dim3 g(EMB / 32, (BATCH * NPTS) / 128);
    gemm_wmma_kernel<2><<<g, 256, 0, stream>>>(agg, Wo, bo, out,
                                               BATCH * NPTS, EMB, EMB);
  }
}
